// UrbanModelV2_15169824489972
// MI455X (gfx1250) — matrix operand order repaired and checked by reading, not compiled, not run on hardware
//
#include <hip/hip_runtime.h>
#include <hip/hip_bf16.h>
#include <math.h>

// ---------------------------------------------------------------------------
// UrbanModelV2 on gfx1250 (CDNA5): wave32 WMMA GEMMs (swizzled B fragments in
// LDS -> vector ds_load_b128) + fp32 L2 atomics for the GCN scatter.
// ---------------------------------------------------------------------------

typedef __attribute__((ext_vector_type(16))) _Float16 v16h;
typedef __attribute__((ext_vector_type(8)))  float    v8f;

#define LN_EPS 1e-5f
#define GWAVES 4   // waves (16-row tiles) per block in GEMM kernels

__device__ __forceinline__ float gelu_f(float x) {
    return 0.5f * x * (1.0f + erff(x * 0.70710678118654752f));
}

// A fragment: 16x32 f16 tile (row m per lane); CDNA5 16-bit A layout:
// lanes 0-15: M=0..15, VGPR0..3 hold K=0..7, VGPR4..7 hold K=16..23
// lanes 16-31: same M, K offset +8. Pairs are contiguous -> 2x ds_load_b128.
__device__ __forceinline__ v16h load_a_frag(const _Float16* A, int lda, int kb, int lane) {
    int m   = lane & 15;
    int khi = (lane >> 4) << 3;
    const _Float16* row = A + m * lda + kb * 32 + khi;
    v16h a;
#pragma unroll
    for (int v = 0; v < 8; ++v) {
        int k = ((v < 4) ? 0 : 16) + ((v & 3) << 1);
        a[2 * v + 0] = row[k + 0];
        a[2 * v + 1] = row[k + 1];
    }
    return a;
}

// Stage W[KDIM x NOUT] (row-major fp32, global) into LDS pre-swizzled in
// B-fragment order: element h of lane `lane` for fragment (kb,nt) sits at
// ((kb*NT+nt)*32 + lane)*16 + h, with k = kb*32 + 16*(h>>3) + 8*(lane>>4) + (h&7) and
// n = (lane&15) + nt*16.  A B-fragment load is then one aligned v16h load.
template <int KDIM, int NOUT>
__device__ __forceinline__ void stage_weights_swz(const float* __restrict__ Wg,
                                                  _Float16* __restrict__ sW,
                                                  int tid, int nthreads) {
    constexpr int NT  = NOUT / 16;
    constexpr int TOT = KDIM * NOUT;        // == (KDIM/32)*NT*512
    for (int idx = tid; idx < TOT; idx += nthreads) {
        int frag = idx >> 9;                // (kb*NT + nt)
        int lane = (idx >> 4) & 31;
        int h    = idx & 15;
        int kb   = frag / NT, nt = frag - kb * NT;
        int k    = kb * 32 + ((h & 8) << 1) + ((lane >> 4) << 3) + (h & 7);
        int n    = (lane & 15) + nt * 16;
        sW[idx]  = (_Float16)Wg[k * NOUT + n];
    }
}

// 16 x NOUT = A[16 x KDIM] * W[KDIM x NOUT], f32 accumulate via v_wmma.
// sW must be in swizzled fragment order (stage_weights_swz).
template <int KDIM, int NT>
__device__ __forceinline__ void wave_gemm(const _Float16* A, const _Float16* sW,
                                          v8f acc[NT], int lane) {
#pragma unroll
    for (int kb = 0; kb < KDIM / 32; ++kb) {
        v16h a = load_a_frag(A, KDIM, kb, lane);
#pragma unroll
        for (int nt = 0; nt < NT; ++nt) {
            v16h b = *(const v16h*)(sW + (((kb * NT + nt) << 9) + (lane << 4)));
            acc[nt] = __builtin_amdgcn_wmma_f32_16x16x32_f16(
                false, a, false, b, (short)0, acc[nt], false, false);
        }
    }
}

// ---------------------------------------------------------------------------
// Encoder: fused[:, :64] = gelu(gelu(LN(ctx@W1+b1))@W2+b2)
//          fused[:,64:96] = gelu((tgt masked)@teW+teb)
// ---------------------------------------------------------------------------
__global__ __launch_bounds__(128) void k_encoder(
    const float* __restrict__ context, const float* __restrict__ target,
    const unsigned char* __restrict__ mask, const float* __restrict__ mask_token,
    const float* __restrict__ ceW1, const float* __restrict__ ceb1,
    const float* __restrict__ ceg,  const float* __restrict__ cebeta,
    const float* __restrict__ ceW2, const float* __restrict__ ceb2,
    const float* __restrict__ teW,  const float* __restrict__ teb,
    float* __restrict__ fused, int n)
{
    __shared__ __align__(32) _Float16 sW1[128 * 64];
    __shared__ __align__(32) _Float16 sW2[64 * 64];
    __shared__ __align__(32) _Float16 sWt[32 * 32];
    __shared__ __align__(32) _Float16 sA[GWAVES][16 * 128];

    stage_weights_swz<128, 64>(ceW1, sW1, threadIdx.x, blockDim.x);
    stage_weights_swz<64, 64>(ceW2, sW2, threadIdx.x, blockDim.x);
    stage_weights_swz<32, 32>(teW, sWt, threadIdx.x, blockDim.x);
    __syncthreads();

    int wave = threadIdx.x >> 5, lane = threadIdx.x & 31;
    int tile = blockIdx.x * GWAVES + wave;
    if (tile * 16 >= n) return;                 // wave-uniform
    int r0 = tile * 16;
    _Float16* Aw = sA[wave];
    int hi = lane >> 4, nlo = lane & 15;

    // stage context tile (fp32 -> f16), coalesced
    for (int i = lane; i < 16 * 128; i += 32) {
        int r = i >> 7, c = i & 127;
        int rr = r0 + r; if (rr >= n) rr = n - 1;
        Aw[i] = (_Float16)context[(size_t)rr * 128 + c];
    }

    // GEMM1: [16x128] x [128x64]
    v8f acc[4] = {};
    wave_gemm<128, 4>(Aw, sW1, acc, lane);

    // +bias, LayerNorm(64), gelu -> stage as A for GEMM2
    float x[4][8];
#pragma unroll
    for (int nt = 0; nt < 4; ++nt) {
        float b = ceb1[nt * 16 + nlo];
#pragma unroll
        for (int r = 0; r < 8; ++r) x[nt][r] = acc[nt][r] + b;
    }
#pragma unroll
    for (int r = 0; r < 8; ++r) {
        float s = x[0][r] + x[1][r] + x[2][r] + x[3][r];
        s += __shfl_xor(s, 1); s += __shfl_xor(s, 2); s += __shfl_xor(s, 4); s += __shfl_xor(s, 8);
        float mu = s * (1.0f / 64.0f);
        float sq = 0.f;
#pragma unroll
        for (int nt = 0; nt < 4; ++nt) { float d = x[nt][r] - mu; sq += d * d; }
        sq += __shfl_xor(sq, 1); sq += __shfl_xor(sq, 2); sq += __shfl_xor(sq, 4); sq += __shfl_xor(sq, 8);
        float rstd = rsqrtf(sq * (1.0f / 64.0f) + LN_EPS);
#pragma unroll
        for (int nt = 0; nt < 4; ++nt) {
            int nc = nt * 16 + nlo;
            float v = (x[nt][r] - mu) * rstd * ceg[nc] + cebeta[nc];
            Aw[(r + 8 * hi) * 64 + nc] = (_Float16)gelu_f(v);
        }
    }

    // GEMM2: [16x64] x [64x64], +bias, gelu -> fused[:, :64]
    v8f acc2[4] = {};
    wave_gemm<64, 4>(Aw, sW2, acc2, lane);
#pragma unroll
    for (int nt = 0; nt < 4; ++nt) {
        int nc = nt * 16 + nlo;
        float b = ceb2[nc];
#pragma unroll
        for (int r = 0; r < 8; ++r) {
            int m = r0 + r + 8 * hi;
            if (m < n) fused[(size_t)m * 96 + nc] = gelu_f(acc2[nt][r] + b);
        }
    }

    // target encoder: mask-mix, GEMM [16x32] x [32x32], +bias, gelu
    for (int i = lane; i < 16 * 32; i += 32) {
        int r = i >> 5, c = i & 31;
        int rr = r0 + r; if (rr >= n) rr = n - 1;
        size_t idx = (size_t)rr * 32 + c;
        float mv = mask[idx] ? 1.0f : 0.0f;
        Aw[i] = (_Float16)(target[idx] * (1.0f - mv) + mask_token[c] * mv);
    }
    v8f acct[2] = {};
    wave_gemm<32, 2>(Aw, sWt, acct, lane);
#pragma unroll
    for (int nt = 0; nt < 2; ++nt) {
        int nc = nt * 16 + nlo;
        float b = teb[nc];
#pragma unroll
        for (int r = 0; r < 8; ++r) {
            int m = r0 + r + 8 * hi;
            if (m < n) fused[(size_t)m * 96 + 64 + nc] = gelu_f(acct[nt][r] + b);
        }
    }
}

// ---------------------------------------------------------------------------
// Y[N,64] = X[N,KDIM] @ W[KDIM,64]   (xW step of GCNConv, no bias)
// ---------------------------------------------------------------------------
template <int KDIM>
__global__ __launch_bounds__(128) void k_gemm_n64(
    const float* __restrict__ X, const float* __restrict__ Wg,
    float* __restrict__ Y, int n)
{
    __shared__ __align__(32) _Float16 sW[KDIM * 64];
    __shared__ __align__(32) _Float16 sA[GWAVES][16 * KDIM];
    stage_weights_swz<KDIM, 64>(Wg, sW, threadIdx.x, blockDim.x);
    __syncthreads();

    int wave = threadIdx.x >> 5, lane = threadIdx.x & 31;
    int tile = blockIdx.x * GWAVES + wave;
    if (tile * 16 >= n) return;
    int r0 = tile * 16;
    _Float16* Aw = sA[wave];
    for (int i = lane; i < 16 * KDIM; i += 32) {
        int r = i / KDIM, c = i % KDIM;
        int rr = r0 + r; if (rr >= n) rr = n - 1;
        Aw[i] = (_Float16)X[(size_t)rr * KDIM + c];
    }
    v8f acc[4] = {};
    wave_gemm<KDIM, 4>(Aw, sW, acc, lane);
    int hi = lane >> 4, nlo = lane & 15;
#pragma unroll
    for (int nt = 0; nt < 4; ++nt)
#pragma unroll
        for (int r = 0; r < 8; ++r) {
            int m = r0 + r + 8 * hi;
            if (m < n) Y[(size_t)m * 64 + nt * 16 + nlo] = acc[nt][r];
        }
}

// ---------------------------------------------------------------------------
// Head: z = concat(h, fused) [N,160]; out = gelu(LN(z@W1+b1))@W2 + b2
// ---------------------------------------------------------------------------
__global__ __launch_bounds__(128) void k_head(
    const float* __restrict__ hsum, const float* __restrict__ fused,
    const float* __restrict__ hW1, const float* __restrict__ hb1,
    const float* __restrict__ hg,  const float* __restrict__ hbeta,
    const float* __restrict__ hW2, const float* __restrict__ hb2,
    float* __restrict__ out, int n)
{
    __shared__ __align__(32) _Float16 sW1[160 * 64];
    __shared__ __align__(32) _Float16 sW2[64 * 32];
    __shared__ __align__(32) _Float16 sA[GWAVES][16 * 160];
    stage_weights_swz<160, 64>(hW1, sW1, threadIdx.x, blockDim.x);
    stage_weights_swz<64, 32>(hW2, sW2, threadIdx.x, blockDim.x);
    __syncthreads();

    int wave = threadIdx.x >> 5, lane = threadIdx.x & 31;
    int tile = blockIdx.x * GWAVES + wave;
    if (tile * 16 >= n) return;
    int r0 = tile * 16;
    _Float16* Aw = sA[wave];
    int hi = lane >> 4, nlo = lane & 15;

    for (int i = lane; i < 16 * 160; i += 32) {
        int r = i / 160, c = i % 160;
        int rr = r0 + r; if (rr >= n) rr = n - 1;
        float v = (c < 64) ? hsum[(size_t)rr * 64 + c]
                           : fused[(size_t)rr * 96 + (c - 64)];
        Aw[i] = (_Float16)v;
    }

    v8f acc[4] = {};
    wave_gemm<160, 4>(Aw, sW1, acc, lane);

    float x[4][8];
#pragma unroll
    for (int nt = 0; nt < 4; ++nt) {
        float b = hb1[nt * 16 + nlo];
#pragma unroll
        for (int r = 0; r < 8; ++r) x[nt][r] = acc[nt][r] + b;
    }
#pragma unroll
    for (int r = 0; r < 8; ++r) {
        float s = x[0][r] + x[1][r] + x[2][r] + x[3][r];
        s += __shfl_xor(s, 1); s += __shfl_xor(s, 2); s += __shfl_xor(s, 4); s += __shfl_xor(s, 8);
        float mu = s * (1.0f / 64.0f);
        float sq = 0.f;
#pragma unroll
        for (int nt = 0; nt < 4; ++nt) { float d = x[nt][r] - mu; sq += d * d; }
        sq += __shfl_xor(sq, 1); sq += __shfl_xor(sq, 2); sq += __shfl_xor(sq, 4); sq += __shfl_xor(sq, 8);
        float rstd = rsqrtf(sq * (1.0f / 64.0f) + LN_EPS);
#pragma unroll
        for (int nt = 0; nt < 4; ++nt) {
            int nc = nt * 16 + nlo;
            float v = (x[nt][r] - mu) * rstd * hg[nc] + hbeta[nc];
            Aw[(r + 8 * hi) * 64 + nc] = (_Float16)gelu_f(v);
        }
    }

    v8f acc2[2] = {};
    wave_gemm<64, 2>(Aw, sW2, acc2, lane);
#pragma unroll
    for (int nt = 0; nt < 2; ++nt) {
        int nc = nt * 16 + nlo;
        float b = hb2[nc];
#pragma unroll
        for (int r = 0; r < 8; ++r) {
            int m = r0 + r + 8 * hi;
            if (m < n) out[(size_t)m * 32 + nc] = acc2[nt][r] + b;
        }
    }
}

// ---------------------------------------------------------------------------
// Graph helpers
// ---------------------------------------------------------------------------
__global__ void k_fill(float* __restrict__ p, long long nElem, float v) {
    long long i = (long long)blockIdx.x * blockDim.x + threadIdx.x;
    long long stride = (long long)gridDim.x * blockDim.x;
    for (; i < nElem; i += stride) p[i] = v;
}

__global__ void k_count(const int* __restrict__ ei, long long E, float* __restrict__ deg) {
    long long t = (long long)blockIdx.x * blockDim.x + threadIdx.x;
    if (t < E) atomicAdd(&deg[ei[E + t]], 1.0f);
}

__global__ void k_finalize_dinv(float* __restrict__ d, int n) {
    int i = blockIdx.x * blockDim.x + threadIdx.x;
    if (i < n) d[i] = rsqrtf(d[i] + 1.0f);
}

// one edge per wave, 2 features per lane (64 features), fp32 L2 atomics
__global__ void k_scatter(const int* __restrict__ ei, long long E,
                          const float* __restrict__ h, const float* __restrict__ dinv,
                          float* __restrict__ agg) {
    long long t = (long long)blockIdx.x * blockDim.x + threadIdx.x;
    long long e = t >> 5;
    if (e >= E) return;
    int lane = (int)(t & 31);
    int src = ei[e], dst = ei[E + e];
    float w = dinv[src] * dinv[dst];
    const float2 v = ((const float2*)(h + (size_t)src * 64))[lane];
    float* ap = agg + (size_t)dst * 64 + lane * 2;
    atomicAdd(ap + 0, v.x * w);
    atomicAdd(ap + 1, v.y * w);
}

__global__ void k_self_gelu(const float* __restrict__ agg, const float* __restrict__ h,
                            const float* __restrict__ dinv, const float* __restrict__ b,
                            float* __restrict__ out, long long total) {
    long long i = (long long)blockIdx.x * blockDim.x + threadIdx.x;
    if (i >= total) return;
    long long node = i >> 6; int f = (int)(i & 63);
    float di = dinv[node];
    out[i] = gelu_f(agg[i] + h[i] * di * di + b[f]);
}

__global__ void k_self_accum(const float* __restrict__ agg, const float* __restrict__ h,
                             const float* __restrict__ dinv, const float* __restrict__ b,
                             const float* __restrict__ alpha, int branch,
                             float* __restrict__ hsum, long long total) {
    long long i = (long long)blockIdx.x * blockDim.x + threadIdx.x;
    if (i >= total) return;
    long long node = i >> 6; int f = (int)(i & 63);
    float a = 1.0f / (1.0f + expf(-alpha[0]));
    float w = branch ? (1.0f - a) : a;
    float di = dinv[node];
    hsum[i] += w * (agg[i] + h[i] * di * di + b[f]);
}

// ---------------------------------------------------------------------------
extern "C" void kernel_launch(void* const* d_in, const int* in_sizes, int n_in,
                              void* d_out, int out_size, void* d_ws, size_t ws_size,
                              hipStream_t stream)
{
    const float* context    = (const float*)d_in[0];
    const float* target     = (const float*)d_in[1];
    const unsigned char* mask = (const unsigned char*)d_in[2];   // jax bool -> 1 byte
    const int*   adj_ei     = (const int*)d_in[3];
    const int*   transit_ei = (const int*)d_in[4];
    const float* mask_token = (const float*)d_in[5];
    const float* ceW1 = (const float*)d_in[6],  *ceb1  = (const float*)d_in[7];
    const float* ceg  = (const float*)d_in[8],  *cebeta= (const float*)d_in[9];
    const float* ceW2 = (const float*)d_in[10], *ceb2  = (const float*)d_in[11];
    const float* teW  = (const float*)d_in[12], *teb   = (const float*)d_in[13];
    const float* g1W  = (const float*)d_in[14], *g1b   = (const float*)d_in[15];
    const float* g2W  = (const float*)d_in[16], *g2b   = (const float*)d_in[17];
    const float* t1W  = (const float*)d_in[18], *t1b   = (const float*)d_in[19];
    const float* t2W  = (const float*)d_in[20], *t2b   = (const float*)d_in[21];
    const float* alpha= (const float*)d_in[22];
    const float* hW1  = (const float*)d_in[23], *hb1   = (const float*)d_in[24];
    const float* hg   = (const float*)d_in[25], *hbeta = (const float*)d_in[26];
    const float* hW2  = (const float*)d_in[27], *hb2   = (const float*)d_in[28];

    const int n = in_sizes[0] / 128;
    const long long E1 = in_sizes[3] / 2;
    const long long E2 = in_sizes[4] / 2;
    const long long nf = (long long)n * 64;

    float* ws     = (float*)d_ws;
    float* dinv_a = ws;  ws += n;
    float* dinv_t = ws;  ws += n;
    float* fused  = ws;  ws += (size_t)n * 96;
    float* hbuf   = ws;  ws += (size_t)n * 64;
    float* aggbuf = ws;  ws += (size_t)n * 64;
    float* h1buf  = ws;  ws += (size_t)n * 64;
    float* hsum   = ws;  ws += (size_t)n * 64;

    auto cdiv = [](long long a, long long b) { return (int)((a + b - 1) / b); };

    // ---- degrees -> dinv ----
    k_fill<<<cdiv(n, 256), 256, 0, stream>>>(dinv_a, n, 0.f);
    k_fill<<<cdiv(n, 256), 256, 0, stream>>>(dinv_t, n, 0.f);
    k_fill<<<cdiv(nf, 256), 256, 0, stream>>>(hsum, nf, 0.f);
    k_count<<<cdiv(E1, 256), 256, 0, stream>>>(adj_ei, E1, dinv_a);
    k_count<<<cdiv(E2, 256), 256, 0, stream>>>(transit_ei, E2, dinv_t);
    k_finalize_dinv<<<cdiv(n, 256), 256, 0, stream>>>(dinv_a, n);
    k_finalize_dinv<<<cdiv(n, 256), 256, 0, stream>>>(dinv_t, n);

    const int tiles   = cdiv(n, 16);
    const int gblocks = cdiv(tiles, GWAVES);

    // ---- encoders -> fused[N,96] ----
    k_encoder<<<gblocks, 128, 0, stream>>>(context, target, mask, mask_token,
        ceW1, ceb1, ceg, cebeta, ceW2, ceb2, teW, teb, fused, n);

    // ---- two GCN branches ----
    for (int branch = 0; branch < 2; ++branch) {
        const int* ei       = branch ? transit_ei : adj_ei;
        const long long E   = branch ? E2 : E1;
        const float* dinv   = branch ? dinv_t : dinv_a;
        const float* W1     = branch ? t1W : g1W;
        const float* b1     = branch ? t1b : g1b;
        const float* W2     = branch ? t2W : g2W;
        const float* b2     = branch ? t2b : g2b;

        k_gemm_n64<96><<<gblocks, 128, 0, stream>>>(fused, W1, hbuf, n);
        k_fill<<<cdiv(nf, 256), 256, 0, stream>>>(aggbuf, nf, 0.f);
        k_scatter<<<cdiv(E * 32, 256), 256, 0, stream>>>(ei, E, hbuf, dinv, aggbuf);
        k_self_gelu<<<cdiv(nf, 256), 256, 0, stream>>>(aggbuf, hbuf, dinv, b1, h1buf, nf);

        k_gemm_n64<64><<<gblocks, 128, 0, stream>>>(h1buf, W2, hbuf, n);
        k_fill<<<cdiv(nf, 256), 256, 0, stream>>>(aggbuf, nf, 0.f);
        k_scatter<<<cdiv(E * 32, 256), 256, 0, stream>>>(ei, E, hbuf, dinv, aggbuf);
        k_self_accum<<<cdiv(nf, 256), 256, 0, stream>>>(aggbuf, hbuf, dinv, b2, alpha,
                                                        branch, hsum, nf);
    }

    // ---- head -> out[N,32] ----
    k_head<<<gblocks, 128, 0, stream>>>(hsum, fused, hW1, hb1, hg, hbeta, hW2, hb2,
                                        (float*)d_out, n);
}
